// TransformerBlock_66666482368603
// MI455X (gfx1250) — matrix operand order repaired and checked by reading, not compiled, not run on hardware
//
#include <hip/hip_runtime.h>

// ---------------------------------------------------------------------------
// Transformer block for MI455X (gfx1250, wave32, WMMA).
// All GEMMs use v_wmma_f32_16x16x32_bf16 (bf16 operands, f32 accumulate).
// Inner loops are software-pipelined with ping-pong register buffers
// (unroll-by-2) so there are no rotation copies and WMMAs overlap the next
// K-step's global loads.
// ---------------------------------------------------------------------------

#define DM     1024
#define NHEADS 16
#define DKH    64
#define SLEN   2048
#define BATCH  4
#define DFF    4096
#define ROWS   (BATCH * SLEN)   // 8192

typedef float  v8f   __attribute__((ext_vector_type(8)));
typedef __bf16 v16bf __attribute__((ext_vector_type(16)));
typedef __bf16 bf16x8 __attribute__((ext_vector_type(8)));
typedef __bf16 bf16x4 __attribute__((ext_vector_type(4)));

union Frag { v16bf v; bf16x8 h[2]; };

__device__ inline v8f wmma_bf16(v16bf a, v16bf b, v8f c) {
  // D = A(16x32 bf16) * B(32x16 bf16) + C(16x16 f32)
  return __builtin_amdgcn_wmma_f32_16x16x32_bf16(false, a, false, b, (short)0, c,
                                                 false, false);
}

// A fragment (16x32, M x K): lane l<16 -> row l, K {0..7, 16..23};
// lane l>=16 -> row l-16, K {8..15, 24..31}.  Two 16B loads per lane.
__device__ inline v16bf load_a_frag(const __bf16* A, int lda, int row0, int k0, int lane) {
  const __bf16* p = A + (size_t)(row0 + (lane & 15)) * lda + k0 + ((lane < 16) ? 0 : 8);
  Frag f;
  f.h[0] = *(const bf16x8*)(p);
  f.h[1] = *(const bf16x8*)(p + 16);
  return f.v;
}

// B fragment (32x16, K x N) from row-major W[N][K] (i.e., B = W^T):
// lane n<16 -> column n, K {0..7, 16..23}; lane n>=16 -> column n-16, K {8..15, 24..31}.
__device__ inline v16bf load_b_frag(const __bf16* W, int ldw, int n0, int k0, int lane) {
  const __bf16* p = W + (size_t)(n0 + (lane & 15)) * ldw + k0 + ((lane < 16) ? 0 : 8);
  Frag f;
  f.h[0] = *(const bf16x8*)(p);
  f.h[1] = *(const bf16x8*)(p + 16);
  return f.v;
}

// ---------------------------------------------------------------------------
// Elementwise / prep kernels
// ---------------------------------------------------------------------------

// n4 = n/4 float4 groups (all tensors here are multiples of 4 elements).
__global__ void cast_f32_bf16_kernel(const float* __restrict__ in,
                                     __bf16* __restrict__ out, int n4) {
  int i = blockIdx.x * blockDim.x + threadIdx.x;
  if (i < n4) {
    float4 v = ((const float4*)in)[i];
    bf16x4 o = {(__bf16)v.x, (__bf16)v.y, (__bf16)v.z, (__bf16)v.w};
    ((bf16x4*)out)[i] = o;
  }
}

// One block per row; 256 threads * 4 floats = 1024 = DM.
__global__ __launch_bounds__(256) void rmsnorm_kernel(const float* __restrict__ x,
                                                      const float* __restrict__ wgt,
                                                      __bf16* __restrict__ out) {
  __shared__ float red[256];
  int row = blockIdx.x, t = threadIdx.x;
  const float4 v = ((const float4*)(x + (size_t)row * DM))[t];
  red[t] = v.x * v.x + v.y * v.y + v.z * v.z + v.w * v.w;
  __syncthreads();
  for (int off = 128; off > 0; off >>= 1) {
    if (t < off) red[t] += red[t + off];
    __syncthreads();
  }
  float rinv = 1.0f / sqrtf(1e-5f + red[0] * (1.0f / (float)DM));
  const float4 wv = ((const float4*)wgt)[t];
  bf16x4 o;
  o[0] = (__bf16)(v.x * rinv * wv.x);
  o[1] = (__bf16)(v.y * rinv * wv.y);
  o[2] = (__bf16)(v.z * rinv * wv.z);
  o[3] = (__bf16)(v.w * rinv * wv.w);
  *(bf16x4*)(out + (size_t)row * DM + t * 4) = o;
}

// RoPE on f32 q/k, write bf16. Layout [B, S, DM], head dim = 64.
__global__ void rope_cast_kernel(const float* __restrict__ in,
                                 __bf16* __restrict__ out, int total) {
  int i = blockIdx.x * blockDim.x + threadIdx.x;
  if (i >= total) return;
  int d  = i & (DM - 1);
  int s  = (i >> 10) & (SLEN - 1);
  int dh = d & (DKH - 1);
  int pair = (dh & 1) ? (i - 1) : (i + 1);
  float sgn = (dh & 1) ? 1.0f : -1.0f;                    // sin sign baked like ref
  float inv = __expf(-(float)(2 * (dh >> 1)) * (9.210340371976184f / 64.0f));
  float ang = (float)s * inv;
  float sn, cs;
  __sincosf(ang, &sn, &cs);
  out[i] = (__bf16)(in[i] * cs + in[pair] * sn * sgn);
}

// v f32 [B,S,DM] -> vt bf16 [B,H,DKH,S]  (so P*V B-fragments are contiguous)
__global__ void vtrans_kernel(const float* __restrict__ vf,
                              __bf16* __restrict__ vt) {
  int i = blockIdx.x * blockDim.x + threadIdx.x;   // B*H*DKH*S = 8388608
  int s = i & (SLEN - 1);
  int d = (i >> 11) & 63;
  int h = (i >> 17) & 15;
  int b = i >> 21;
  vt[i] = (__bf16)vf[(size_t)(b * SLEN + s) * DM + h * DKH + d];
}

// ---------------------------------------------------------------------------
// Generic WMMA GEMM: C[M,N] = A[M,K](bf16) * W[N,K]^T(bf16), f32 out.
// EPI==1: out = resid + acc. Wave tile 32x64 (2 A-frags x 4 B-frags,
// 8 accumulators). Block = 8 waves -> 64x256 tile. Ping-pong pipelined.
// Requires K % 64 == 0 (holds: K = 1024 or 4096).
// ---------------------------------------------------------------------------
__device__ inline void load_a2(const __bf16* A, int K, int row0, int k, int lane,
                               v16bf (&d)[2]) {
  d[0] = load_a_frag(A, K, row0, k, lane);
  d[1] = load_a_frag(A, K, row0 + 16, k, lane);
}
__device__ inline void load_b4(const __bf16* W, int K, int col0, int k, int lane,
                               v16bf (&d)[4]) {
#pragma unroll
  for (int j = 0; j < 4; ++j) d[j] = load_b_frag(W, K, col0 + j * 16, k, lane);
}
__device__ inline void mma24(v8f (&acc)[2][4], const v16bf (&a)[2], const v16bf (&b)[4]) {
#pragma unroll
  for (int j = 0; j < 4; ++j) {
    acc[0][j] = wmma_bf16(a[0], b[j], acc[0][j]);
    acc[1][j] = wmma_bf16(a[1], b[j], acc[1][j]);
  }
}

template <int EPI>
__global__ __launch_bounds__(256) void gemm_kernel(const __bf16* __restrict__ A,
                                                   const __bf16* __restrict__ W,
                                                   const float* __restrict__ resid,
                                                   float* __restrict__ out,
                                                   int M, int N, int K) {
  int lane = threadIdx.x & 31, w = threadIdx.x >> 5;
  int row0 = (blockIdx.x * 2 + (w >> 2)) * 32;
  int col0 = (blockIdx.y * 4 + (w & 3)) * 64;
  v8f acc[2][4] = {};
  v16bf aA[2], aB[2], bA[4], bB[4];

  load_a2(A, K, row0, 0, lane, aA);
  load_b4(W, K, col0, 0, lane, bA);
  int k0 = 0;
  for (; k0 + 64 < K; k0 += 64) {
    load_a2(A, K, row0, k0 + 32, lane, aB);
    load_b4(W, K, col0, k0 + 32, lane, bB);
    mma24(acc, aA, bA);
    load_a2(A, K, row0, k0 + 64, lane, aA);
    load_b4(W, K, col0, k0 + 64, lane, bA);
    mma24(acc, aB, bB);
  }
  load_a2(A, K, row0, K - 32, lane, aB);
  load_b4(W, K, col0, K - 32, lane, bB);
  mma24(acc, aA, bA);
  mma24(acc, aB, bB);

  int n = lane & 15, hi = (lane >> 4) * 8;
#pragma unroll
  for (int m = 0; m < 2; ++m)
#pragma unroll
    for (int j = 0; j < 4; ++j)
#pragma unroll
      for (int r = 0; r < 8; ++r) {
        size_t idx = (size_t)(row0 + m * 16 + hi + r) * N + col0 + j * 16 + n;
        float v = acc[m][j][r];
        if (EPI == 1) v += resid[idx];
        out[idx] = v;
      }
}

// Fused w1/w3 GEMM + SwiGLU epilogue, bf16 output [M, DFF].
// Wave tile 16x32 per weight matrix; ping-pong pipelined.
__global__ __launch_bounds__(256) void gemm_w13_kernel(const __bf16* __restrict__ A,
                                                       const __bf16* __restrict__ W1,
                                                       const __bf16* __restrict__ W3,
                                                       __bf16* __restrict__ out,
                                                       int M, int N, int K) {
  int lane = threadIdx.x & 31, w = threadIdx.x >> 5;
  int row0 = (blockIdx.x * 2 + (w >> 2)) * 16;
  int col0 = (blockIdx.y * 4 + (w & 3)) * 32;
  v8f a1[2] = {}, a3[2] = {};
  v16bf aA, aB, b1A[2], b3A[2], b1B[2], b3B[2];

  auto loadset = [&](v16bf& a, v16bf (&b1)[2], v16bf (&b3)[2], int k) {
    a = load_a_frag(A, K, row0, k, lane);
#pragma unroll
    for (int j = 0; j < 2; ++j) {
      b1[j] = load_b_frag(W1, K, col0 + j * 16, k, lane);
      b3[j] = load_b_frag(W3, K, col0 + j * 16, k, lane);
    }
  };
  auto compute = [&](const v16bf& a, const v16bf (&b1)[2], const v16bf (&b3)[2]) {
#pragma unroll
    for (int j = 0; j < 2; ++j) {
      a1[j] = wmma_bf16(a, b1[j], a1[j]);
      a3[j] = wmma_bf16(a, b3[j], a3[j]);
    }
  };

  loadset(aA, b1A, b3A, 0);
  int k0 = 0;
  for (; k0 + 64 < K; k0 += 64) {
    loadset(aB, b1B, b3B, k0 + 32);
    compute(aA, b1A, b3A);
    loadset(aA, b1A, b3A, k0 + 64);
    compute(aB, b1B, b3B);
  }
  loadset(aB, b1B, b3B, K - 32);
  compute(aA, b1A, b3A);
  compute(aB, b1B, b3B);

  int n = lane & 15, hi = (lane >> 4) * 8;
#pragma unroll
  for (int j = 0; j < 2; ++j)
#pragma unroll
    for (int r = 0; r < 8; ++r) {
      float p1 = a1[j][r], p3 = a3[j][r];
      float v = p1 / (1.0f + __expf(-p1)) * p3;   // silu(p1) * p3
      out[(size_t)(row0 + hi + r) * N + col0 + j * 16 + n] = (__bf16)v;
    }
}

// ---------------------------------------------------------------------------
// Causal attention: one workgroup per (b, h, 16-row q tile). 8 waves.
// Dynamic LDS: sc (16x2048 f32 scores) + red (256 f32) + cbuf (2x16x64 f32).
// ---------------------------------------------------------------------------
#define ATTN_SMEM (16 * SLEN * 4 + 256 * 4 + 2 * 16 * 64 * 4)  // 140288 B

__global__ __launch_bounds__(256) void attn_kernel(const __bf16* __restrict__ qb,
                                                   const __bf16* __restrict__ kb,
                                                   const __bf16* __restrict__ vt,
                                                   __bf16* __restrict__ ctx) {
  extern __shared__ char smem[];
  float* sc   = (float*)smem;                       // [16][2048]
  float* red  = (float*)(smem + 16 * SLEN * 4);     // [256]
  float* cbuf = red + 256;                          // [2][16][64]

  int tid = threadIdx.x, lane = tid & 31, w = tid >> 5;
  int qt = blockIdx.x, h = blockIdx.y, b = blockIdx.z;
  int q0 = qt * 16;
  int nkt = qt + 1;                                 // causal: key tiles 0..qt

  // --- Phase 1: scores = (Q K^T) / 8, masked, into LDS ---
  v16bf fqa[2];
  {
    int r = q0 + (lane & 15);
    int kbo = (lane < 16) ? 0 : 8;
    const __bf16* qp = qb + (size_t)(b * SLEN + r) * DM + h * DKH;
#pragma unroll
    for (int c = 0; c < 2; ++c) {
      Frag f;
      f.h[0] = *(const bf16x8*)(qp + c * 32 + kbo);
      f.h[1] = *(const bf16x8*)(qp + c * 32 + kbo + 16);
      fqa[c] = f.v;
    }
  }
  {
    const size_t kbase = (size_t)(b * SLEN) * DM + h * DKH + ((lane < 16) ? 0 : 8);
    auto load_k = [&](v16bf (&d)[2], int t) {
      const __bf16* kp = kb + kbase + (size_t)(t * 16 + (lane & 15)) * DM;
#pragma unroll
      for (int c = 0; c < 2; ++c) {
        Frag f;
        f.h[0] = *(const bf16x8*)(kp + c * 32);
        f.h[1] = *(const bf16x8*)(kp + c * 32 + 16);
        d[c] = f.v;
      }
    };
    auto score_tile = [&](const v16bf (&fk)[2], int t) {
      v8f acc = {};
      acc = wmma_bf16(fqa[0], fk[0], acc);
      acc = wmma_bf16(fqa[1], fk[1], acc);
      int kk0 = t * 16;
      int n = lane & 15, hi = (lane >> 4) * 8;
#pragma unroll
      for (int r = 0; r < 8; ++r) {
        int lr = r + hi;
        int kkg = kk0 + n;
        sc[lr * SLEN + kkg] = (kkg <= q0 + lr) ? acc[r] * 0.125f : -3.0e38f;
      }
    };

    int t = w;
    v16bf kA[2], kB[2];
    if (t < nkt) load_k(kA, t);
    for (; t + 16 < nkt; t += 16) {
      load_k(kB, t + 8);
      score_tile(kA, t);
      load_k(kA, t + 16);
      score_tile(kB, t + 8);
    }
    if (t < nkt) {
      if (t + 8 < nkt) {
        load_k(kB, t + 8);
        score_tile(kA, t);
        score_tile(kB, t + 8);
      } else {
        score_tile(kA, t);
      }
    }
  }
  __syncthreads();

  // --- Phase 2: softmax over each of 16 rows (16 threads per row) ---
  int row = tid >> 4, sub = tid & 15;
  int L = q0 + row + 1;
  float m = -3.0e38f;
  for (int j = sub; j < L; j += 16) m = fmaxf(m, sc[row * SLEN + j]);
  red[tid] = m;
  __syncthreads();
  float rm = red[row * 16];
#pragma unroll
  for (int i = 1; i < 16; ++i) rm = fmaxf(rm, red[row * 16 + i]);
  __syncthreads();
  float s = 0.0f;
  for (int j = sub; j < L; j += 16) {
    float e = __expf(sc[row * SLEN + j] - rm);
    sc[row * SLEN + j] = e;
    s += e;
  }
  red[tid] = s;
  __syncthreads();
  float rs = 0.0f;
#pragma unroll
  for (int i = 0; i < 16; ++i) rs += red[row * 16 + i];
  float rinv = 1.0f / rs;
  int kmax = (nkt * 16 + 31) & ~31;                 // pad to WMMA K granularity
  for (int j = sub; j < kmax; j += 16)
    sc[row * SLEN + j] = (j < L) ? sc[row * SLEN + j] * rinv : 0.0f;
  __syncthreads();

  // --- Phase 3: ctx = P * V. Wave = (d-tile 0..3) x (K-half 0..1) ---
  int d0 = (w & 3) * 16;
  int half = w >> 2;
  v8f cacc = {};
  {
    const __bf16* vbase = vt + ((size_t)(b * NHEADS + h) * DKH + d0 + (lane & 15)) * SLEN +
                          ((lane < 16) ? 0 : 8);
    auto load_v = [&](v16bf& d, int kkc) {
      Frag f;
      f.h[0] = *(const bf16x8*)(vbase + kkc);
      f.h[1] = *(const bf16x8*)(vbase + kkc + 16);
      d = f.v;
    };
    auto pv = [&](const v16bf& fv, int kkc) {
      int r = lane & 15;
      int kbo = (lane < 16) ? 0 : 8;
      Frag fa;
      const float* pp = sc + r * SLEN + kkc + kbo;
#pragma unroll
      for (int i = 0; i < 8; ++i) fa.h[0][i] = (__bf16)pp[i];
#pragma unroll
      for (int i = 0; i < 8; ++i) fa.h[1][i] = (__bf16)pp[16 + i];
      cacc = wmma_bf16(fa.v, fv, cacc);
    };

    int kkc = half * 32;                 // stride 64 per wave-half
    v16bf vA, vB;
    if (kkc < kmax) load_v(vA, kkc);
    for (; kkc + 128 < kmax; kkc += 128) {
      load_v(vB, kkc + 64);
      pv(vA, kkc);
      load_v(vA, kkc + 128);
      pv(vB, kkc + 64);
    }
    if (kkc < kmax) {
      if (kkc + 64 < kmax) {
        load_v(vB, kkc + 64);
        pv(vA, kkc);
        pv(vB, kkc + 64);
      } else {
        pv(vA, kkc);
      }
    }
  }
  {
    int n = lane & 15, hi = (lane >> 4) * 8;
#pragma unroll
    for (int r = 0; r < 8; ++r)
      cbuf[half * 1024 + (r + hi) * 64 + d0 + n] = cacc[r];
  }
  __syncthreads();

  // --- Combine halves, write bf16 ctx [B,S,DM] ---
  {
    int i0 = tid * 4;
    int lr = i0 >> 6, d = i0 & 63;
    float4 p0 = *(const float4*)(cbuf + i0);
    float4 p1 = *(const float4*)(cbuf + 1024 + i0);
    bf16x4 o;
    o[0] = (__bf16)(p0.x + p1.x);
    o[1] = (__bf16)(p0.y + p1.y);
    o[2] = (__bf16)(p0.z + p1.z);
    o[3] = (__bf16)(p0.w + p1.w);
    *(bf16x4*)(ctx + (size_t)(b * SLEN + q0 + lr) * DM + h * DKH + d) = o;
  }
}

// ---------------------------------------------------------------------------
// Host-side orchestration
// ---------------------------------------------------------------------------
#define MB ((size_t)1024 * 1024)

extern "C" void kernel_launch(void* const* d_in, const int* in_sizes, int n_in,
                              void* d_out, int out_size, void* d_ws, size_t ws_size,
                              hipStream_t stream) {
  const float* x     = (const float*)d_in[0];
  const float* wq    = (const float*)d_in[1];
  const float* wk    = (const float*)d_in[2];
  const float* wv    = (const float*)d_in[3];
  const float* wo    = (const float*)d_in[4];
  const float* n1w   = (const float*)d_in[5];
  const float* n2w   = (const float*)d_in[6];
  const float* w1    = (const float*)d_in[7];
  const float* w2    = (const float*)d_in[8];
  const float* w3    = (const float*)d_in[9];
  float* out = (float*)d_out;
  char* ws = (char*)d_ws;

  // Workspace layout (deterministic reuse), ~208 MB total:
  __bf16* hb   = (__bf16*)(ws + 0);          // 16 MB  rmsnorm1 output
  __bf16* wqb  = (__bf16*)(ws + 16 * MB);    //  2 MB
  __bf16* wkb  = (__bf16*)(ws + 18 * MB);
  __bf16* wvb  = (__bf16*)(ws + 20 * MB);
  __bf16* wob  = (__bf16*)(ws + 22 * MB);
  __bf16* w1b  = (__bf16*)(ws + 24 * MB);    //  8 MB
  __bf16* w3b  = (__bf16*)(ws + 32 * MB);
  __bf16* w2b  = (__bf16*)(ws + 40 * MB);
  float*  qf   = (float*)(ws + 48 * MB);     // 32 MB (reused as y)
  float*  kf   = (float*)(ws + 80 * MB);     // 32 MB (reused as h2b)
  float*  vf   = (float*)(ws + 112 * MB);    // 32 MB
  __bf16* qbb  = (__bf16*)(ws + 144 * MB);   // 16 MB (reused by ppb)
  __bf16* kbb  = (__bf16*)(ws + 160 * MB);   // 16 MB
  __bf16* vt   = (__bf16*)(ws + 176 * MB);   // 16 MB
  __bf16* ctxb = (__bf16*)(ws + 192 * MB);   // 16 MB
  float*  y    = qf;                         // attn residual output (f32)
  __bf16* h2b  = (__bf16*)kf;                // rmsnorm2 output (bf16)
  __bf16* ppb  = (__bf16*)(ws + 144 * MB);   // 64 MB swiglu output, reuses q/k/vt/ctx

  const int NSQ = DM * DM;    // 1048576
  const int NFF = DFF * DM;   // 4194304
  const int NACT = ROWS * DM; // 8388608

  // 1) weights -> bf16 (vectorized x4)
  cast_f32_bf16_kernel<<<NSQ / 4 / 256, 256, 0, stream>>>(wq, wqb, NSQ / 4);
  cast_f32_bf16_kernel<<<NSQ / 4 / 256, 256, 0, stream>>>(wk, wkb, NSQ / 4);
  cast_f32_bf16_kernel<<<NSQ / 4 / 256, 256, 0, stream>>>(wv, wvb, NSQ / 4);
  cast_f32_bf16_kernel<<<NSQ / 4 / 256, 256, 0, stream>>>(wo, wob, NSQ / 4);
  cast_f32_bf16_kernel<<<NFF / 4 / 256, 256, 0, stream>>>(w1, w1b, NFF / 4);
  cast_f32_bf16_kernel<<<NFF / 4 / 256, 256, 0, stream>>>(w2, w2b, NFF / 4);
  cast_f32_bf16_kernel<<<NFF / 4 / 256, 256, 0, stream>>>(w3, w3b, NFF / 4);

  // 2) h = rmsnorm(x) (bf16)
  rmsnorm_kernel<<<ROWS, 256, 0, stream>>>(x, n1w, hb);

  // 3) QKV projections (WMMA), f32 outputs. Block tile 64x256.
  dim3 gqkv(ROWS / 64, DM / 256);
  gemm_kernel<0><<<gqkv, 256, 0, stream>>>(hb, wqb, nullptr, qf, ROWS, DM, DM);
  gemm_kernel<0><<<gqkv, 256, 0, stream>>>(hb, wkb, nullptr, kf, ROWS, DM, DM);
  gemm_kernel<0><<<gqkv, 256, 0, stream>>>(hb, wvb, nullptr, vf, ROWS, DM, DM);

  // 4) RoPE(q), RoPE(k) -> bf16 ; v -> vt bf16 [B,H,DKH,S]
  rope_cast_kernel<<<NACT / 256, 256, 0, stream>>>(qf, qbb, NACT);
  rope_cast_kernel<<<NACT / 256, 256, 0, stream>>>(kf, kbb, NACT);
  vtrans_kernel<<<NACT / 256, 256, 0, stream>>>(vf, vt);

  // 5) causal attention -> ctx bf16
  dim3 gattn(SLEN / 16, NHEADS, BATCH);
  attn_kernel<<<gattn, 256, ATTN_SMEM, stream>>>(qbb, kbb, vt, ctxb);

  // 6) y = x + ctx @ wo^T
  gemm_kernel<1><<<dim3(ROWS / 64, DM / 256), 256, 0, stream>>>(ctxb, wob, x, y,
                                                                ROWS, DM, DM);

  // 7) h2 = rmsnorm(y)
  rmsnorm_kernel<<<ROWS, 256, 0, stream>>>(y, n2w, h2b);

  // 8) pp = silu(h2 @ w1^T) * (h2 @ w3^T)  (fused, bf16, wave tile 16x32)
  gemm_w13_kernel<<<dim3(ROWS / 32, DFF / 128), 256, 0, stream>>>(h2b, w1b, w3b, ppb,
                                                                  ROWS, DFF, DM);

  // 9) out = y + pp @ w2^T
  gemm_kernel<1><<<dim3(ROWS / 64, DM / 256), 256, 0, stream>>>(ppb, w2b, y, out,
                                                                ROWS, DM, DFF);
}
